// CrossAttention_11613591568496
// MI455X (gfx1250) — matrix operand order repaired and checked by reading, not compiled, not run on hardware
//
#include <hip/hip_runtime.h>

// ---------------- problem constants ----------------
constexpr int Bsz   = 8;
constexpr int NQ    = 256;
constexpr int Tt    = 32;
constexpr int Mkv   = 256;
constexpr int TM    = Tt * Mkv;     // 8192
constexpr int DIM   = 768;
constexpr int KVD   = 1024;
constexpr int H     = 8;
constexpr int DH    = 64;
constexpr int INNER = H * DH;       // 512

typedef __attribute__((ext_vector_type(16))) __bf16 v16bf;
typedef __attribute__((ext_vector_type(8)))  __bf16 bf16x8;
typedef __attribute__((ext_vector_type(8)))  float  v8f;
typedef __attribute__((ext_vector_type(4)))  int    v4i;

// ---------------- async global->LDS copy (CDNA5 ASYNCcnt path) ----------------
#if __has_builtin(__builtin_amdgcn_global_load_async_to_lds_b128)
#define ASYNC_LDS 1
__device__ inline void cp_async16(const __bf16* g, __bf16* l) {
    __builtin_amdgcn_global_load_async_to_lds_b128(
        (__attribute__((address_space(1))) v4i*)g,
        (__attribute__((address_space(3))) v4i*)l, 0, 0);
}
#else
#define ASYNC_LDS 0
__device__ inline void cp_async16(const __bf16* g, __bf16* l) {
    *(bf16x8*)l = *(const bf16x8*)g;            // sync fallback
}
#endif

template<int N>
__device__ inline void cp_wait() {
#if ASYNC_LDS
#if __has_builtin(__builtin_amdgcn_s_wait_asynccnt)
    __builtin_amdgcn_s_wait_asynccnt(N);
#else
    asm volatile("s_wait_asynccnt %0" :: "n"(N) : "memory");
#endif
#endif
}

// Build a 16-element bf16 A/B fragment from two 16B-aligned LDS chunks
// (two ds_load_b128 per fragment).
__device__ inline v16bf load_frag(const __bf16* lo, const __bf16* hi) {
    bf16x8 l = *(const bf16x8*)lo;
    bf16x8 h = *(const bf16x8*)hi;
    v16bf r;
#pragma unroll
    for (int i = 0; i < 8; ++i) { r[i] = l[i]; r[i + 8] = h[i]; }
    return r;
}

__device__ inline v8f wmma_bf16(v16bf a, v16bf b, v8f c) {
    return __builtin_amdgcn_wmma_f32_16x16x32_bf16(false, a, false, b, (short)0, c,
                                                   false, false);
}

// ---------------- LayerNorm (f32 in -> bf16 out), one row per block ----------------
__global__ void ln_bf16_kernel(const float* __restrict__ x, const float* __restrict__ g,
                               const float* __restrict__ b, __bf16* __restrict__ out,
                               int width) {
    int row = blockIdx.x;
    const float* xr = x + (size_t)row * width;
    __bf16* orow = out + (size_t)row * width;
    __shared__ float red[256];
    float s = 0.f, s2 = 0.f;
    for (int i = threadIdx.x; i < width; i += 256) { float v = xr[i]; s += v; s2 += v * v; }
    red[threadIdx.x] = s; __syncthreads();
    for (int st = 128; st > 0; st >>= 1) {
        if (threadIdx.x < st) red[threadIdx.x] += red[threadIdx.x + st];
        __syncthreads();
    }
    float mean = red[0] / width; __syncthreads();
    red[threadIdx.x] = s2; __syncthreads();
    for (int st = 128; st > 0; st >>= 1) {
        if (threadIdx.x < st) red[threadIdx.x] += red[threadIdx.x + st];
        __syncthreads();
    }
    float var  = red[0] / width - mean * mean;
    float rstd = rsqrtf(var + 1e-5f);
    for (int i = threadIdx.x; i < width; i += 256) {
        float v = (xr[i] - mean) * rstd * g[i] + b[i];
        orow[i] = (__bf16)v;
    }
}

// ---------------- weight convert + transpose: W[K][N] f32 -> Wt[N][K] bf16 ----------------
__global__ void cvt_transpose_bf16(const float* __restrict__ W, __bf16* __restrict__ Wt,
                                   int K, int N) {
    int n = blockIdx.x * 16 + (threadIdx.x & 15);
    int k = blockIdx.y * 16 + (threadIdx.x >> 4);
    if (n < N && k < K) Wt[(size_t)n * K + k] = (__bf16)W[(size_t)k * N + n];
}

// ---------------- bf16 WMMA GEMM: C[M,N] = A[M,K] @ Bt[N,K]^T ----------------
// Block tile 128(M) x 64(N), 128 threads = 4 waves, wave = 32M x 64N (8 accumulators).
// Double-buffered LDS, async tile copies (6 x b128 per thread per K-step).
// All of M%128==0, N%64==0, K%32==0 hold for every call here.
template<bool BF16OUT>
__global__ void gemm_bf16_wmma(const __bf16* __restrict__ A, const __bf16* __restrict__ Bt,
                               __bf16* __restrict__ Cbf, float* __restrict__ Cf,
                               int Mrows, int N, int K) {
    __shared__ __align__(16) __bf16 As[2][128 * 32];   // [m][k] row-major
    __shared__ __align__(16) __bf16 Bs[2][64 * 32];    // [n][k] N-major (B pre-transposed)
    int t = threadIdx.x;
    int lane = t & 31, wave = t >> 5;
    int laneHi = (lane >> 4) & 1, lm = lane & 15;
    int bm = blockIdx.y * 128, bn = blockIdx.x * 64;
    v8f acc[2][4] = {};

    int arow[4], ack[4];
#pragma unroll
    for (int it = 0; it < 4; ++it) { int idx = t + it * 128; arow[it] = idx >> 2; ack[it] = (idx & 3) * 8; }
    int brow[2], bck[2];
#pragma unroll
    for (int it = 0; it < 2; ++it) { int idx = t + it * 128; brow[it] = idx >> 2; bck[it] = (idx & 3) * 8; }

    auto issue = [&](int buf, int k0) {
#pragma unroll
        for (int it = 0; it < 4; ++it)
            cp_async16(&A[(size_t)(bm + arow[it]) * K + k0 + ack[it]],
                       &As[buf][arow[it] * 32 + ack[it]]);
#pragma unroll
        for (int it = 0; it < 2; ++it)
            cp_async16(&Bt[(size_t)(bn + brow[it]) * K + k0 + bck[it]],
                       &Bs[buf][brow[it] * 32 + bck[it]]);
    };

    issue(0, 0);
    int nsteps = K / 32;
    for (int s = 0; s < nsteps; ++s) {
        int buf = s & 1;
        if (s + 1 < nsteps) { issue(buf ^ 1, (s + 1) * 32); cp_wait<6>(); }
        else                { cp_wait<0>(); }
        __syncthreads();

        const __bf16* a0 = &As[buf][(wave * 32 + lm) * 32];
        v16bf af0 = load_frag(a0 + laneHi * 8, a0 + laneHi * 8 + 16);
        const __bf16* a1 = a0 + 16 * 32;
        v16bf af1 = load_frag(a1 + laneHi * 8, a1 + laneHi * 8 + 16);
#pragma unroll
        for (int nt = 0; nt < 4; ++nt) {
            const __bf16* br = &Bs[buf][(nt * 16 + lm) * 32];
            v16bf bfrag = load_frag(br + laneHi * 8, br + laneHi * 8 + 16);
            acc[0][nt] = wmma_bf16(af0, bfrag, acc[0][nt]);
            acc[1][nt] = wmma_bf16(af1, bfrag, acc[1][nt]);
        }
        __syncthreads();
    }
#pragma unroll
    for (int ms = 0; ms < 2; ++ms)
#pragma unroll
        for (int nt = 0; nt < 4; ++nt)
#pragma unroll
            for (int r = 0; r < 8; ++r) {
                int gm = bm + wave * 32 + ms * 16 + r + laneHi * 8;
                int gn = bn + nt * 16 + lm;
                float v = acc[ms][nt][r];
                if constexpr (BF16OUT) Cbf[(size_t)gm * N + gn] = (__bf16)v;
                else                   Cf [(size_t)gm * N + gn] = v;
            }
}

// ---------------- flash attention: per (b,h), 64 q-rows per block ----------------
// qb  : bf16 [B*NQ][INNER]
// kvp : bf16 [B*TM][2*INNER]  (cols [h*64..] = K head h, cols [INNER + h*64..] = V)
// outb: bf16 [B*NQ][INNER]
__global__ void attn_wmma_kernel(const __bf16* __restrict__ qb,
                                 const __bf16* __restrict__ kvp,
                                 __bf16* __restrict__ outb) {
    __shared__ __align__(16) __bf16 Qs[64 * 64];   // [qm][dh]
    __shared__ __align__(16) __bf16 Ks[64 * 64];   // [tm][dh] == N-major for S=QK^T
    __shared__ __align__(16) __bf16 Vt[64 * 64];   // [dh][tm] == N-major for O=PV
    __shared__ __align__(16) float  Ss[64 * 64];
    __shared__ __align__(16) __bf16 Ps[64 * 64];
    __shared__ float mrow[64], lrow[64], arow_s[64], red2[128];

    int t = threadIdx.x, lane = t & 31, wave = t >> 5;
    int laneHi = (lane >> 4) & 1, lm = lane & 15;
    int b = blockIdx.y / H, h = blockIdx.y % H;
    int q0 = blockIdx.x * 64;
    const float scale = 0.125f;   // DH^-0.5

#pragma unroll
    for (int it = 0; it < 4; ++it) {               // Q tile (async, resident whole kernel)
        int idx = t + it * 128;
        int row = idx >> 3, c = (idx & 7) * 8;
        cp_async16(&qb[(size_t)(b * NQ + q0 + row) * INNER + h * DH + c],
                   &Qs[row * 64 + c]);
    }
    if (t < 64) { mrow[t] = -1e30f; lrow[t] = 0.f; }
    v8f oacc[4] = {};
    cp_wait<0>();
    __syncthreads();

    for (int tm0 = 0; tm0 < TM; tm0 += 64) {
#pragma unroll
        for (int it = 0; it < 4; ++it) {           // K tile async; V tile sync + transpose
            int idx = t + it * 128;
            int row = idx >> 3, c = (idx & 7) * 8;
            size_t base = (size_t)(b * TM + tm0 + row) * (2 * INNER);
            cp_async16(&kvp[base + h * DH + c], &Ks[row * 64 + c]);
            bf16x8 vv = *(const bf16x8*)&kvp[base + INNER + h * DH + c];
#pragma unroll
            for (int j = 0; j < 8; ++j) Vt[(c + j) * 64 + row] = vv[j];
        }
        cp_wait<0>();
        __syncthreads();

        // S = Q K^T  (K-dim = DH = 64 -> 2 wmma K-steps, 4 N-subtiles)
        v8f sacc[4] = {};
#pragma unroll
        for (int ks = 0; ks < 2; ++ks) {
            const __bf16* ar = &Qs[(wave * 16 + lm) * 64 + ks * 32];
            v16bf af = load_frag(ar + laneHi * 8, ar + laneHi * 8 + 16);
#pragma unroll
            for (int nt = 0; nt < 4; ++nt) {
                const __bf16* br = &Ks[(nt * 16 + lm) * 64 + ks * 32];
                v16bf bf_ = load_frag(br + laneHi * 8, br + laneHi * 8 + 16);
                sacc[nt] = wmma_bf16(af, bf_, sacc[nt]);
            }
        }
#pragma unroll
        for (int nt = 0; nt < 4; ++nt)
#pragma unroll
            for (int r = 0; r < 8; ++r)
                Ss[(wave * 16 + r + laneHi * 8) * 64 + nt * 16 + lm] = sacc[nt][r] * scale;
        __syncthreads();

        // online softmax: 2 threads per row, 32 columns each
        {
            int r = t >> 1, half = t & 1;
            const float* srow = &Ss[r * 64 + half * 32];
            float mx = -1e30f;
#pragma unroll 8
            for (int j = 0; j < 32; ++j) mx = fmaxf(mx, srow[j]);
            red2[t] = mx;
            __syncthreads();
            float mold = mrow[r];
            float mnew = fmaxf(mold, fmaxf(red2[2 * r], red2[2 * r + 1]));
            float alpha = __expf(mold - mnew);
            float sum = 0.f;
            __bf16* prow = &Ps[r * 64 + half * 32];
#pragma unroll 8
            for (int j = 0; j < 32; ++j) {
                float p = __expf(srow[j] - mnew);
                sum += p;
                prow[j] = (__bf16)p;
            }
            red2[t] = sum;
            __syncthreads();
            if (half == 0) {
                lrow[r] = lrow[r] * alpha + red2[2 * r] + red2[2 * r + 1];
                mrow[r] = mnew;
                arow_s[r] = alpha;
            }
        }
        __syncthreads();

        // O = O*alpha + P @ V
#pragma unroll
        for (int r = 0; r < 8; ++r) {
            float a = arow_s[wave * 16 + r + laneHi * 8];
#pragma unroll
            for (int nt = 0; nt < 4; ++nt) oacc[nt][r] *= a;
        }
#pragma unroll
        for (int ks = 0; ks < 2; ++ks) {
            const __bf16* ar = &Ps[(wave * 16 + lm) * 64 + ks * 32];
            v16bf af = load_frag(ar + laneHi * 8, ar + laneHi * 8 + 16);
#pragma unroll
            for (int nt = 0; nt < 4; ++nt) {
                const __bf16* br = &Vt[(nt * 16 + lm) * 64 + ks * 32];
                v16bf bf_ = load_frag(br + laneHi * 8, br + laneHi * 8 + 16);
                oacc[nt] = wmma_bf16(af, bf_, oacc[nt]);
            }
        }
        __syncthreads();
    }

#pragma unroll
    for (int nt = 0; nt < 4; ++nt)
#pragma unroll
        for (int r = 0; r < 8; ++r) {
            int m = wave * 16 + r + laneHi * 8;
            float v = oacc[nt][r] / lrow[m];
            outb[(size_t)(b * NQ + q0 + m) * INNER + h * DH + nt * 16 + lm] = (__bf16)v;
        }
}

// ---------------- host orchestration ----------------
extern "C" void kernel_launch(void* const* d_in, const int* in_sizes, int n_in,
                              void* d_out, int out_size, void* d_ws, size_t ws_size,
                              hipStream_t stream) {
    (void)in_sizes; (void)n_in; (void)out_size; (void)ws_size;
    const float* x    = (const float*)d_in[0];
    const float* k_v  = (const float*)d_in[1];
    const float* g_q  = (const float*)d_in[2];
    const float* b_q  = (const float*)d_in[3];
    const float* g_kv = (const float*)d_in[4];
    const float* b_kv = (const float*)d_in[5];
    const float* Wq   = (const float*)d_in[6];
    const float* Wkv  = (const float*)d_in[7];
    const float* Wo   = (const float*)d_in[8];
    float* out = (float*)d_out;

    char* ws = (char*)d_ws;
    size_t off = 0;
    auto alloc = [&](size_t bytes) -> void* {
        void* p = ws + off;
        off = (off + bytes + 255) & ~(size_t)255;
        return p;
    };
    __bf16* xln    = (__bf16*)alloc((size_t)Bsz * NQ * DIM * 2);
    __bf16* kvln   = (__bf16*)alloc((size_t)Bsz * TM * KVD * 2);
    __bf16* Wq_t   = (__bf16*)alloc((size_t)DIM * INNER * 2);        // [INNER][DIM]
    __bf16* Wkv_t  = (__bf16*)alloc((size_t)KVD * 2 * INNER * 2);    // [2*INNER][KVD]
    __bf16* Wo_t   = (__bf16*)alloc((size_t)INNER * DIM * 2);        // [DIM][INNER]
    __bf16* qb     = (__bf16*)alloc((size_t)Bsz * NQ * INNER * 2);
    __bf16* kvp    = (__bf16*)alloc((size_t)Bsz * TM * 2 * INNER * 2);
    __bf16* attnO  = (__bf16*)alloc((size_t)Bsz * NQ * INNER * 2);

    // 1) LayerNorms (fused f32->bf16)
    ln_bf16_kernel<<<Bsz * NQ, 256, 0, stream>>>(x, g_q, b_q, xln, DIM);
    ln_bf16_kernel<<<Bsz * TM, 256, 0, stream>>>(k_v, g_kv, b_kv, kvln, KVD);

    // 2) weight convert + transpose (B operands become [N][K] bf16)
    cvt_transpose_bf16<<<dim3(INNER / 16, DIM / 16), 256, 0, stream>>>(Wq, Wq_t, DIM, INNER);
    cvt_transpose_bf16<<<dim3((2 * INNER) / 16, KVD / 16), 256, 0, stream>>>(Wkv, Wkv_t, KVD, 2 * INNER);
    cvt_transpose_bf16<<<dim3(DIM / 16, INNER / 16), 256, 0, stream>>>(Wo, Wo_t, INNER, DIM);

    // 3) q = xln @ Wq : [2048 x 768] @ [768 x 512] -> bf16
    gemm_bf16_wmma<true><<<dim3(INNER / 64, (Bsz * NQ) / 128), 128, 0, stream>>>(
        xln, Wq_t, qb, nullptr, Bsz * NQ, INNER, DIM);

    // 4) kvp = kvln @ Wkv : [65536 x 1024] @ [1024 x 1024] -> bf16 (dominant GEMM)
    gemm_bf16_wmma<true><<<dim3((2 * INNER) / 64, (Bsz * TM) / 128), 128, 0, stream>>>(
        kvln, Wkv_t, kvp, nullptr, Bsz * TM, 2 * INNER, KVD);

    // 5) flash attention per (b,h), 64 q-rows / block
    attn_wmma_kernel<<<dim3(NQ / 64, Bsz * H), 128, 0, stream>>>(qb, kvp, attnO);

    // 6) out = attnO @ Wo : [2048 x 512] @ [512 x 768] -> f32
    gemm_bf16_wmma<false><<<dim3(DIM / 64, (Bsz * NQ) / 128), 128, 0, stream>>>(
        attnO, Wo_t, nullptr, out, Bsz * NQ, DIM, INNER);
}
